// MyQwen3SparseMLP_36661840838793
// MI455X (gfx1250) — matrix operand order repaired and checked by reading, not compiled, not run on hardware
//
#include <hip/hip_runtime.h>

// ---------------- problem constants ----------------
#define T_TOK    4096
#define HID      1024
#define BDIM_    64
#define DIN_     1088          // HID + BDIM
#define INTER_   2048
#define NEXP     10            // E_TOTAL
#define TPAD_    5376          // 4096 + 10*128 worst-case padding
#define MAXTILES 42            // >= sum ceil(cnt_e/128)
#define NK1      (DIN_ / 32)   // 34 K-steps, GEMM1
#define NK2      (INTER_ / 32) // 64 K-steps, GEMM2

// ---------------- WMMA types ----------------
typedef __attribute__((ext_vector_type(16))) __bf16 v16bf;
typedef __attribute__((ext_vector_type(8)))  float  v8f;

union AFrag { uint4 q[2]; v16bf v; };
union BFPair { __bf16 h[2]; unsigned u; };   // packed bf16 pair -> one b32 store

static __device__ __forceinline__ v8f wmma_bf16(const AFrag& a, const AFrag& b, v8f c) {
  // D = A(16x32 bf16) * B(32x16 bf16) + C(16x16 f32)
  return __builtin_amdgcn_wmma_f32_16x16x32_bf16(false, a.v, false, b.v,
                                                 (short)0, c, false, false);
}

// ---------------- TDM (Tensor Data Mover) helpers ----------------
typedef __attribute__((ext_vector_type(4))) unsigned int tdm_v4u;
typedef __attribute__((ext_vector_type(8))) int          tdm_v8i;
typedef __attribute__((ext_vector_type(4))) int          tdm_v4i;

// Wave-relative LDS byte offset: flat LDS aperture keeps the offset in addr[31:0].
static __device__ __forceinline__ unsigned lds_off(const void* p) {
  return (unsigned)(uintptr_t)p;
}

// D# group 0: count=1, lds_addr, global_addr[56:0], type=2 ("image")
static __device__ __forceinline__ tdm_v4u tdm_g0(unsigned lds, const void* g) {
  unsigned long long ga = (unsigned long long)(uintptr_t)g;
  tdm_v4u g0;
  g0[0] = 1u;                                             // count=1, user mode
  g0[1] = lds;                                            // lds_addr (bytes)
  g0[2] = (unsigned)ga;                                   // global_addr[31:0]
  g0[3] = (unsigned)((ga >> 32) & 0x01ffffffu) | 0x80000000u; // ga[56:32] | type=2<<30
  return g0;
}

// D# group 1 for a dense 2-D bf16 tile: tensor_dim == tile_dim (fully interior tiles)
static __device__ __forceinline__ tdm_v8i tdm_g1_2d(unsigned d0, unsigned d1,
                                                    unsigned stride_elems) {
  tdm_v8i g1;
  g1[0] = 0x00010000;              // workgroup_mask=0 | data_size=1 (2 bytes)
  g1[1] = (int)(d0 << 16);         // atomic_barrier_addr=0 | tensor_dim0[15:0]
  g1[2] = (int)(d1 << 16);         // tensor_dim0[31:16]=0 | tensor_dim1[15:0]
  g1[3] = (int)(d0 << 16);         // tensor_dim1[31:16]=0 | tile_dim0
  g1[4] = (int)(d1 & 0xffffu);     // tile_dim1 | tile_dim2=0
  g1[5] = (int)stride_elems;       // tensor_dim0_stride[31:0]
  g1[6] = 0;                       // stride0 hi | tensor_dim1_stride lo (unused, 2D)
  g1[7] = 0;
  return g1;
}

static __device__ __forceinline__ void tdm_load2d(unsigned lds, const void* g,
                                                  unsigned d0, unsigned d1,
                                                  unsigned stride_elems) {
  tdm_v4i z4 = {0, 0, 0, 0};
  tdm_v8i z8 = {0, 0, 0, 0, 0, 0, 0, 0};
  __builtin_amdgcn_tensor_load_to_lds(tdm_g0(lds, g), tdm_g1_2d(d0, d1, stride_elems),
                                      z4, z4, z8, 0);
}
static __device__ __forceinline__ void tdm_store2d(unsigned lds, void* g,
                                                   unsigned d0, unsigned d1,
                                                   unsigned stride_elems) {
  tdm_v4i z4 = {0, 0, 0, 0};
  tdm_v8i z8 = {0, 0, 0, 0, 0, 0, 0, 0};
  __builtin_amdgcn_tensor_store_from_lds(tdm_g0(lds, g), tdm_g1_2d(d0, d1, stride_elems),
                                         z4, z4, z8, 0);
}

// ---------------- workspace layout (bytes) ----------------
#define WS_CNT    0u
#define WS_CUR    64u
#define WS_OFFS   128u
#define WS_NT     192u
#define WS_TE     256u
#define WS_TR     512u
#define WS_PERM   768u
#define WS_TOKPOS (WS_PERM + TPAD_ * 4u)
#define WS_XB     ((WS_TOKPOS + T_TOK * 4u + 4095u) & ~4095u)
#define WS_H      ((WS_XB + TPAD_ * DIN_ * 2u + 4095u) & ~4095u)

// ---------------- routing ----------------
__global__ void k_init(int* smallz, uint4* xb, int nxb4) {
  if (blockIdx.x == 0 && threadIdx.x < 48) smallz[threadIdx.x] = 0;
  int stride = gridDim.x * blockDim.x;
  for (int j = blockIdx.x * blockDim.x + threadIdx.x; j < nxb4; j += stride)
    xb[j] = uint4{0u, 0u, 0u, 0u};
}

static __device__ __forceinline__ int expert_of(int pos, int act) {
  int e = 3 * (act - 1) + pos;
  return e < 0 ? 0 : e;
}

__global__ void k_route(const int* __restrict__ pos, const int* __restrict__ act,
                        int* __restrict__ cnt) {
  int i = blockIdx.x * blockDim.x + threadIdx.x;
  if (i >= T_TOK) return;
  atomicAdd(&cnt[expert_of(pos[i], act[i])], 1);
}

__global__ void k_scan(const int* __restrict__ cnt, int* __restrict__ offs,
                       int* __restrict__ te, int* __restrict__ tr,
                       int* __restrict__ nt) {
  if (threadIdx.x != 0) return;
  int o = 0, n = 0;
  for (int e = 0; e < NEXP; ++e) {
    offs[e] = o;
    int tiles = (cnt[e] + 127) >> 7;
    for (int t = 0; t < tiles; ++t) { te[n] = e; tr[n] = o + t * 128; ++n; }
    o += tiles * 128;
  }
  offs[NEXP] = o;
  *nt = n;
}

__global__ void k_scatter(const int* __restrict__ pos, const int* __restrict__ act,
                          const int* __restrict__ offs, int* __restrict__ cur,
                          int* __restrict__ perm, int* __restrict__ tokpos) {
  int i = blockIdx.x * blockDim.x + threadIdx.x;
  if (i >= T_TOK) return;
  int e = expert_of(pos[i], act[i]);
  int p = offs[e] + atomicAdd(&cur[e], 1);
  perm[p] = i;
  tokpos[i] = p;
}

__global__ void k_gather(const float* __restrict__ hs, const float* __restrict__ bemb,
                         const int* __restrict__ beh, const int* __restrict__ tokpos,
                         __bf16* __restrict__ Xb) {
  int i = blockIdx.x;
  int p = tokpos[i];
  int b = beh[i];
  for (int c = threadIdx.x * 4; c < DIN_; c += blockDim.x * 4) {
    float4 f;
    if (c < HID) f = *(const float4*)(hs + (size_t)i * HID + c);
    else         f = *(const float4*)(bemb + (size_t)b * BDIM_ + (c - HID));
    union { __bf16 h[4]; uint2 u; } cv;
    cv.h[0] = (__bf16)f.x; cv.h[1] = (__bf16)f.y;
    cv.h[2] = (__bf16)f.z; cv.h[3] = (__bf16)f.w;
    *(uint2*)(Xb + (size_t)p * DIN_ + c) = cv.u;
  }
}

// ---------------- GEMM1: [128 x DIN] x (Wg|Wu [DIN x 64]) -> silu*mul -> H bf16
__global__ __launch_bounds__(256) void k_gemm1(
    const float* __restrict__ Wg, const float* __restrict__ Wu,
    const __bf16* __restrict__ Xb, __bf16* __restrict__ H,
    const int* __restrict__ te, const int* __restrict__ tr,
    const int* __restrict__ ntp) {
  if ((int)blockIdx.y >= *ntp) return;
  const int e    = te[blockIdx.y];
  const int row0 = tr[blockIdx.y];
  const int n0   = blockIdx.x * 64;

  const float* wg = Wg + (size_t)e * DIN_ * INTER_;
  const float* wu = Wu + (size_t)e * DIN_ * INTER_;

  __shared__ __bf16 As[2][128 * 32];   // [row][k]   (TDM-filled)
  __shared__ __bf16 Bg[2][64 * 32];    // transposed [n][k]
  __shared__ __bf16 Bu[2][64 * 32];

  const int tid = threadIdx.x;
  const int wave = tid >> 5, lane = tid & 31;
  const int wm = wave >> 1, wn = wave & 1;      // 4x2 waves -> 128x64 tile
  const int half = lane >> 4, l15 = lane & 15;

  // B stage: thread owns k-pair (2kp, 2kp+1) x 4 consecutive n
  const int kp = tid >> 4;               // 0..15 -> k = 2*kp
  const int nb = (tid & 15) * 4;         // 0..60
  float4 rgA, rgB, ruA, ruB;             // rows 2kp and 2kp+1

  // per-wave TDM slice: 16 rows x 32 k of the A tile
  auto tdmA = [&](int buf, int ks) {
    tdm_load2d(lds_off(&As[buf][0]) + (unsigned)wave * 16u * 32u * 2u,
               Xb + (size_t)(row0 + wave * 16) * DIN_ + ks * 32,
               /*d0=*/32u, /*d1=*/16u, /*stride=*/DIN_);
  };
  auto loadB = [&](int ks) {
    const float* g0p = wg + (size_t)(ks * 32 + 2 * kp) * INTER_ + n0 + nb;
    rgA = *(const float4*)(g0p);
    rgB = *(const float4*)(g0p + INTER_);
    const float* u0p = wu + (size_t)(ks * 32 + 2 * kp) * INTER_ + n0 + nb;
    ruA = *(const float4*)(u0p);
    ruB = *(const float4*)(u0p + INTER_);
  };
  auto storeB = [&](int buf) {
    float ga[4] = {rgA.x, rgA.y, rgA.z, rgA.w};
    float gb[4] = {rgB.x, rgB.y, rgB.z, rgB.w};
    float ua[4] = {ruA.x, ruA.y, ruA.z, ruA.w};
    float ub[4] = {ruB.x, ruB.y, ruB.z, ruB.w};
#pragma unroll
    for (int i = 0; i < 4; ++i) {
      BFPair pg, pu;
      pg.h[0] = (__bf16)ga[i]; pg.h[1] = (__bf16)gb[i];   // -> v_cvt_pk_bf16_f32
      pu.h[0] = (__bf16)ua[i]; pu.h[1] = (__bf16)ub[i];
      *(unsigned*)&Bg[buf][(nb + i) * 32 + 2 * kp] = pg.u;  // one ds_store_b32
      *(unsigned*)&Bu[buf][(nb + i) * 32 + 2 * kp] = pu.u;
    }
  };

  v8f z = {0.f, 0.f, 0.f, 0.f, 0.f, 0.f, 0.f, 0.f};
  v8f ag[2][2] = {{z, z}, {z, z}}, au[2][2] = {{z, z}, {z, z}};

  auto compute = [&](int buf) {
    AFrag fa[2], fg[2], fu[2];
#pragma unroll
    for (int mi = 0; mi < 2; ++mi) {
      const __bf16* p = &As[buf][(wm * 32 + mi * 16 + l15) * 32];
      fa[mi].q[0] = *(const uint4*)(p + 8 * half);
      fa[mi].q[1] = *(const uint4*)(p + 16 + 8 * half);
    }
#pragma unroll
    for (int ni = 0; ni < 2; ++ni) {
      const __bf16* pg = &Bg[buf][(wn * 32 + ni * 16 + l15) * 32 + 8 * half];
      fg[ni].q[0] = *(const uint4*)(pg);
      fg[ni].q[1] = *(const uint4*)(pg + 16);
      const __bf16* pu = &Bu[buf][(wn * 32 + ni * 16 + l15) * 32 + 8 * half];
      fu[ni].q[0] = *(const uint4*)(pu);
      fu[ni].q[1] = *(const uint4*)(pu + 16);
    }
#pragma unroll
    for (int mi = 0; mi < 2; ++mi)
#pragma unroll
      for (int ni = 0; ni < 2; ++ni) {
        ag[mi][ni] = wmma_bf16(fa[mi], fg[ni], ag[mi][ni]);
        au[mi][ni] = wmma_bf16(fa[mi], fu[ni], au[mi][ni]);
      }
  };

  tdmA(0, 0);
  loadB(0);
  storeB(0);
  __builtin_amdgcn_s_wait_tensorcnt(0);
  __syncthreads();
  for (int ks = 0; ks < NK1; ++ks) {
    int buf = ks & 1, nxt = buf ^ 1;
    if (ks + 1 < NK1) { tdmA(nxt, ks + 1); loadB(ks + 1); }
    compute(buf);
    if (ks + 1 < NK1) {
      storeB(nxt);
      __builtin_amdgcn_s_wait_tensorcnt(0);
      __syncthreads();
    }
  }

  // epilogue: h = silu(g)*u -> LDS tile -> TDM store to H (coalesced DMA)
  __syncthreads();                       // all WMMAs done; A buffers are free
  __bf16* Ct = (__bf16*)&As[0][0];       // 128 x 64 bf16 = 16 KB
#pragma unroll
  for (int mi = 0; mi < 2; ++mi)
#pragma unroll
    for (int ni = 0; ni < 2; ++ni) {
      int col = wn * 32 + ni * 16 + l15;
#pragma unroll
      for (int v = 0; v < 8; ++v) {
        int r = wm * 32 + mi * 16 + v + 8 * half;
        float g = ag[mi][ni][v], u = au[mi][ni][v];
        float h = g / (1.f + __expf(-g)) * u;
        Ct[r * 64 + col] = (__bf16)h;
      }
    }
  __syncthreads();
  tdm_store2d(lds_off(Ct) + (unsigned)wave * 16u * 64u * 2u,
              H + (size_t)(row0 + wave * 16) * INTER_ + n0,
              /*d0=*/64u, /*d1=*/16u, /*stride=*/INTER_);
  __builtin_amdgcn_s_wait_tensorcnt(0);  // S_ENDPGM would also wait-idle
}

// ---------------- GEMM2: H[128 x 2048] x Wd[2048 x 128] -> scatter to out
__global__ __launch_bounds__(256) void k_gemm2(
    const float* __restrict__ Wd, const __bf16* __restrict__ H,
    float* __restrict__ out,
    const int* __restrict__ te, const int* __restrict__ tr,
    const int* __restrict__ ntp, const int* __restrict__ offs,
    const int* __restrict__ cnt, const int* __restrict__ perm) {
  if ((int)blockIdx.y >= *ntp) return;
  const int e    = te[blockIdx.y];
  const int row0 = tr[blockIdx.y];
  const int n0   = blockIdx.x * 128;

  const float* wd = Wd + (size_t)e * INTER_ * HID;

  __shared__ __bf16 As[2][128 * 32];   // [row][k]   (TDM-filled)
  __shared__ __bf16 Bt[2][128 * 32];   // transposed [n][k]

  const int tid = threadIdx.x;
  const int wave = tid >> 5, lane = tid & 31;
  const int wm = wave >> 1, wn = wave & 1;      // 4x2 waves -> 128x128 tile (wave: 32x64)
  const int half = lane >> 4, l15 = lane & 15;

  // B stage: thread owns k-pair (2kp, 2kp+1) x 8 consecutive n
  const int kp = tid >> 4;               // 0..15
  const int nb = (tid & 15) * 8;         // 0..120
  float4 rbA[2], rbB[2];

  auto tdmA = [&](int buf, int ks) {
    tdm_load2d(lds_off(&As[buf][0]) + (unsigned)wave * 16u * 32u * 2u,
               H + (size_t)(row0 + wave * 16) * INTER_ + ks * 32,
               /*d0=*/32u, /*d1=*/16u, /*stride=*/INTER_);
  };
  auto loadB = [&](int ks) {
    const float* p0 = wd + (size_t)(ks * 32 + 2 * kp) * HID + n0 + nb;
    rbA[0] = *(const float4*)(p0);
    rbA[1] = *(const float4*)(p0 + 4);
    rbB[0] = *(const float4*)(p0 + HID);
    rbB[1] = *(const float4*)(p0 + HID + 4);
  };
  auto storeB = [&](int buf) {
    float a[8] = {rbA[0].x, rbA[0].y, rbA[0].z, rbA[0].w,
                  rbA[1].x, rbA[1].y, rbA[1].z, rbA[1].w};
    float b[8] = {rbB[0].x, rbB[0].y, rbB[0].z, rbB[0].w,
                  rbB[1].x, rbB[1].y, rbB[1].z, rbB[1].w};
#pragma unroll
    for (int i = 0; i < 8; ++i) {
      BFPair p;
      p.h[0] = (__bf16)a[i]; p.h[1] = (__bf16)b[i];       // -> v_cvt_pk_bf16_f32
      *(unsigned*)&Bt[buf][(nb + i) * 32 + 2 * kp] = p.u; // one ds_store_b32
    }
  };

  v8f z = {0.f, 0.f, 0.f, 0.f, 0.f, 0.f, 0.f, 0.f};
  v8f acc[2][4] = {{z, z, z, z}, {z, z, z, z}};

  auto compute = [&](int buf) {
    AFrag fa[2], fb[4];
#pragma unroll
    for (int mi = 0; mi < 2; ++mi) {
      const __bf16* p = &As[buf][(wm * 32 + mi * 16 + l15) * 32];
      fa[mi].q[0] = *(const uint4*)(p + 8 * half);
      fa[mi].q[1] = *(const uint4*)(p + 16 + 8 * half);
    }
#pragma unroll
    for (int ni = 0; ni < 4; ++ni) {
      const __bf16* p = &Bt[buf][(wn * 64 + ni * 16 + l15) * 32 + 8 * half];
      fb[ni].q[0] = *(const uint4*)(p);
      fb[ni].q[1] = *(const uint4*)(p + 16);
    }
#pragma unroll
    for (int mi = 0; mi < 2; ++mi)
#pragma unroll
      for (int ni = 0; ni < 4; ++ni)
        acc[mi][ni] = wmma_bf16(fa[mi], fb[ni], acc[mi][ni]);
  };

  tdmA(0, 0);
  loadB(0);
  storeB(0);
  __builtin_amdgcn_s_wait_tensorcnt(0);
  __syncthreads();
  for (int ks = 0; ks < NK2; ++ks) {
    int buf = ks & 1, nxt = buf ^ 1;
    if (ks + 1 < NK2) { tdmA(nxt, ks + 1); loadB(ks + 1); }
    compute(buf);
    if (ks + 1 < NK2) {
      storeB(nxt);
      __builtin_amdgcn_s_wait_tensorcnt(0);
      __syncthreads();
    }
  }

  const int lim = offs[e] + cnt[e];   // mask padding rows
#pragma unroll
  for (int mi = 0; mi < 2; ++mi)
#pragma unroll
    for (int ni = 0; ni < 4; ++ni) {
      int col = n0 + wn * 64 + ni * 16 + l15;
#pragma unroll
      for (int v = 0; v < 8; ++v) {
        int r = row0 + wm * 32 + mi * 16 + v + 8 * half;
        if (r < lim) out[(size_t)perm[r] * HID + col] = acc[mi][ni][v];
      }
    }
}

// ---------------- host launcher ----------------
extern "C" void kernel_launch(void* const* d_in, const int* in_sizes, int n_in,
                              void* d_out, int out_size, void* d_ws, size_t ws_size,
                              hipStream_t stream) {
  const float* hs   = (const float*)d_in[0];
  const int*   pos  = (const int*)d_in[1];
  const int*   beh  = (const int*)d_in[2];
  const int*   act  = (const int*)d_in[3];
  const float* bemb = (const float*)d_in[4];
  const float* Wg   = (const float*)d_in[5];
  const float* Wu   = (const float*)d_in[6];
  const float* Wd   = (const float*)d_in[7];
  float* out = (float*)d_out;

  char* ws = (char*)d_ws;
  int* cnt    = (int*)(ws + WS_CNT);
  int* cur    = (int*)(ws + WS_CUR);
  int* offs   = (int*)(ws + WS_OFFS);
  int* nt     = (int*)(ws + WS_NT);
  int* te     = (int*)(ws + WS_TE);
  int* tr     = (int*)(ws + WS_TR);
  int* perm   = (int*)(ws + WS_PERM);
  int* tokpos = (int*)(ws + WS_TOKPOS);
  __bf16* Xb  = (__bf16*)(ws + WS_XB);
  __bf16* H   = (__bf16*)(ws + WS_H);

  const int nxb4 = (TPAD_ * DIN_ * 2) / 16;
  k_init<<<2048, 256, 0, stream>>>(cnt, (uint4*)Xb, nxb4);
  k_route<<<(T_TOK + 255) / 256, 256, 0, stream>>>(pos, act, cnt);
  k_scan<<<1, 32, 0, stream>>>(cnt, offs, te, tr, nt);
  k_scatter<<<(T_TOK + 255) / 256, 256, 0, stream>>>(pos, act, offs, cur, perm, tokpos);
  k_gather<<<T_TOK, 128, 0, stream>>>(hs, bemb, beh, tokpos, Xb);
  k_gemm1<<<dim3(INTER_ / 64, MAXTILES), 256, 0, stream>>>(Wg, Wu, Xb, H, te, tr, nt);
  k_gemm2<<<dim3(HID / 128, MAXTILES), 256, 0, stream>>>(Wd, H, out, te, tr, nt, offs, cnt, perm);
}
